// DynamicContextModule2d_56633438765359
// MI455X (gfx1250) — matrix operand order repaired and checked by reading, not compiled, not run on hardware
//
#include <hip/hip_runtime.h>

typedef __attribute__((ext_vector_type(16))) __bf16        v16bf;
typedef __attribute__((ext_vector_type(8)))  float         v8f;
typedef __attribute__((ext_vector_type(4)))  unsigned int  u32x4;
typedef __attribute__((ext_vector_type(8)))  unsigned int  u32x8;
typedef __attribute__((ext_vector_type(4)))  float         f32x4;

#define NB   32
#define CIN  512
#define COUT 512
#define HH   56
#define WWID 56
#define HWD  (HH * WWID)   // 3136
#define KSZ  7
#define PP   (KSZ * KSZ)   // 49

__device__ __forceinline__ unsigned short f2bf(float f) {
    unsigned int u = __float_as_uint(f);
    u += 0x7FFFu + ((u >> 16) & 1u);   // round-to-nearest-even
    return (unsigned short)(u >> 16);
}

// -------------------------------------------------------------------------
// C[z][m][n] = sum_k A_bf16[m][k] * B_f32[z][k][n] + bias[m]
// A: [512][512] bf16 row-major (TDM -> LDS), B: [NB][512][3136] f32,
// C: [NB][512][3136] f32.  Block: 128 threads = 4 waves (2x2),
// tile BM=128, BN=64, BK=32.  A tile is DMA'd by the Tensor Data Mover:
// each wave owns a 32-row slab (TDM ignores EXEC, so issue per wave).
// -------------------------------------------------------------------------
__global__ __launch_bounds__(128)
void gemm_wmma_bf16(const unsigned short* __restrict__ Abf,
                    const float* __restrict__ B,
                    const float* __restrict__ bias,
                    float* __restrict__ C) {
    __shared__ __align__(64) unsigned short As[128 * 32];  // [m][k] 8 KB
    __shared__ __align__(64) unsigned short Bs[64 * 32];   // [n][k] 4 KB

    const int t    = threadIdx.x;
    const int lane = t & 31;
    const int wv   = __builtin_amdgcn_readfirstlane(t >> 5);  // uniform wave id
    const int wm   = wv >> 1;        // 0..1 -> 64 rows each
    const int wn   = wv & 1;         // 0..1 -> 32 cols each
    const int n0   = blockIdx.x * 64;
    const int m0   = blockIdx.y * 128;
    const int z    = blockIdx.z;

    const float* Bz = B + (size_t)z * CIN * HWD;

    // ---- constant part of Tensor DMA descriptor (D# group 1) ----
    // data_size=1 (2B), tensor_dim0=512, tensor_dim1=512, tile_dim0=32,
    // tile_dim1=32 (one wave's slab), tensor_dim0_stride=512
    u32x8 g1;
    g1[0] = 0x00010000u;   // workgroup_mask=0, data_size=1 (2 bytes)
    g1[1] = 0x02000000u;   // tensor_dim0[15:0]=512 in bits[63:48]
    g1[2] = 0x02000000u;   // tensor_dim1[15:0]=512 in bits[95:80]
    g1[3] = 0x00200000u;   // tile_dim0=32 in bits[127:112]
    g1[4] = 0x00000020u;   // tile_dim1=32, tile_dim2=0
    g1[5] = 512u;          // tensor_dim0_stride low
    g1[6] = 0u;
    g1[7] = 0u;

    const unsigned ldsA = (unsigned)(unsigned long long)(const void*)As + (unsigned)wv * 2048u;
    const unsigned long long abase =
        (unsigned long long)(const void*)Abf + ((size_t)(m0 + wv * 32) * CIN) * 2ull;

    v8f acc[4][2];
#pragma unroll
    for (int i = 0; i < 4; ++i)
#pragma unroll
        for (int j = 0; j < 2; ++j) acc[i][j] = 0.0f;

    const int rowSel = lane & 15;
    const int hiSel  = lane >> 4;

    for (int k0 = 0; k0 < CIN; k0 += 32) {
        // ---- A tile via Tensor Data Mover: 32 rows x 32 bf16 per wave ----
        {
            unsigned long long gaddr = abase + (unsigned)k0 * 2u;
            u32x4 g0;
            g0[0] = 1u;                                   // count=1, user D#
            g0[1] = ldsA;                                 // LDS byte address
            g0[2] = (unsigned)gaddr;                      // global_addr[31:0]
            g0[3] = (unsigned)((gaddr >> 32) & 0x01FFFFFFu) | (2u << 30); // addr[56:32], type=2
            asm volatile("tensor_load_to_lds %0, %1" :: "s"(g0), "s"(g1) : "memory");
        }
        // ---- stage B tile 32x64 fp32 -> bf16, transposed to [n][k],
        //      packed pairs (k,k+1) -> one b32 DS store ----
#pragma unroll
        for (int i = 0; i < 2; ++i) {
            int idx = i * 128 + t;              // 0..255 pair-tasks
            int kp = idx >> 4, f4 = idx & 15;   // kp: k-pair 0..15, f4: n/4 0..15
            const float* r0 = Bz + (size_t)(k0 + 2 * kp) * HWD + n0 + f4 * 4;
            f32x4 va = *(const f32x4*)r0;
            f32x4 vb = *(const f32x4*)(r0 + HWD);
            if (k0 + 32 < CIN) {                // L2 prefetch of next K tile
                __builtin_prefetch(r0 + 32 * HWD, 0, 1);
            }
#pragma unroll
            for (int j = 0; j < 4; ++j) {
                unsigned pk = (unsigned)f2bf(va[j]) | ((unsigned)f2bf(vb[j]) << 16);
                *(unsigned*)&Bs[(f4 * 4 + j) * 32 + 2 * kp] = pk;
            }
        }
        __builtin_amdgcn_s_wait_tensorcnt(0);   // own slab of A landed in LDS
        __syncthreads();                        // all waves' slabs + B visible

        // ---- load fragments per ISA 16-bit layouts ----
        // A 16x32: lane = M row; VGPR0-3: K = hiSel*8 .. +7 ; VGPR4-7: +16
        v16bf afrag[4];
#pragma unroll
        for (int i = 0; i < 4; ++i) {
            int row = wm * 64 + i * 16 + rowSel;
            u32x4 lo = *(const u32x4*)&As[row * 32 + hiSel * 8];
            u32x4 hi = *(const u32x4*)&As[row * 32 + hiSel * 8 + 16];
            u32x8 w  = __builtin_shufflevector(lo, hi, 0, 1, 2, 3, 4, 5, 6, 7);
            afrag[i] = __builtin_bit_cast(v16bf, w);
        }
        // B 32x16: lane = N col; VGPR0-3: K = hiSel*8 .. +7 ; VGPR4-7: +16
        v16bf bfrag[2];
#pragma unroll
        for (int j = 0; j < 2; ++j) {
            int col = wn * 32 + j * 16 + rowSel;
            u32x4 lo = *(const u32x4*)&Bs[col * 32 + hiSel * 8];
            u32x4 hi = *(const u32x4*)&Bs[col * 32 + hiSel * 8 + 16];
            u32x8 w  = __builtin_shufflevector(lo, hi, 0, 1, 2, 3, 4, 5, 6, 7);
            bfrag[j] = __builtin_bit_cast(v16bf, w);
        }

#pragma unroll
        for (int i = 0; i < 4; ++i)
#pragma unroll
            for (int j = 0; j < 2; ++j)
                acc[i][j] = __builtin_amdgcn_wmma_f32_16x16x32_bf16(
                    false, afrag[i], false, bfrag[j],
                    (short)0, acc[i][j], false, false);
        __syncthreads();
    }

    // ---- epilogue: C/D layout: VGPR r -> M = r + 8*hiSel, N = lane&15 ----
#pragma unroll
    for (int i = 0; i < 4; ++i)
#pragma unroll
        for (int j = 0; j < 2; ++j) {
            int nc = n0 + wn * 32 + j * 16 + rowSel;
#pragma unroll
            for (int r = 0; r < 8; ++r) {
                int mg = m0 + wm * 64 + i * 16 + 8 * hiSel + r;
                C[((size_t)z * COUT + mg) * HWD + nc] = acc[i][j][r] + bias[mg];
            }
        }
}

// -------------------------------------------------------------------------
// 8x8 exact average pool: x[N,CIN,56,56] -> pooled[N,CIN,7,7]
// -------------------------------------------------------------------------
__global__ void pool_kernel(const float* __restrict__ x, float* __restrict__ pooled) {
    int idx = blockIdx.x * blockDim.x + threadIdx.x;
    if (idx >= NB * CIN * PP) return;
    int p = idx % PP, nc = idx / PP;
    int py = p / KSZ, px = p % KSZ;
    const float* base = x + (size_t)nc * HWD + (py * 8) * WWID + px * 8;
    float s = 0.f;
#pragma unroll
    for (int r = 0; r < 8; ++r) {
        f32x4 v0 = *(const f32x4*)(base + r * WWID);
        f32x4 v1 = *(const f32x4*)(base + r * WWID + 4);
        s += v0[0] + v0[1] + v0[2] + v0[3] + v1[0] + v1[1] + v1[2] + v1[3];
    }
    pooled[idx] = s * (1.0f / 64.0f);
}

// -------------------------------------------------------------------------
// down[n,o,p] = b_down[o] + sum_c w_down[o,c] * pooled[n,c,p]   (tiny GEMM)
// -------------------------------------------------------------------------
__global__ void down_kernel(const float* __restrict__ pooled,
                            const float* __restrict__ w,
                            const float* __restrict__ bias,
                            float* __restrict__ down) {
    int idx = blockIdx.x * blockDim.x + threadIdx.x;
    if (idx >= NB * COUT * PP) return;
    int p = idx % PP;
    int o = (idx / PP) % COUT;
    int n = idx / (PP * COUT);
    const float* wr = w + (size_t)o * CIN;
    const float* pr = pooled + (size_t)n * CIN * PP + p;
    float s = bias[o];
    for (int c = 0; c < CIN; ++c) s += wr[c] * pr[(size_t)c * PP];
    down[idx] = s;
}

// -------------------------------------------------------------------------
// Dynamic depthwise 7x7, pad 3: one block per (c, n); channel image in LDS
// -------------------------------------------------------------------------
__global__ __launch_bounds__(256)
void dwconv_kernel(const float* __restrict__ up,
                   const float* __restrict__ down,
                   float* __restrict__ out) {
    __shared__ float tile[HWD];
    __shared__ float ker[PP];
    const int c = blockIdx.x, n = blockIdx.y;
    const int t = threadIdx.x;
    const float* src = up + ((size_t)n * COUT + c) * HWD;
    for (int p = t; p < HWD; p += 256) tile[p] = src[p];
    if (t < PP) ker[t] = down[((size_t)n * COUT + c) * PP + t];
    __syncthreads();
    float* dst = out + ((size_t)n * COUT + c) * HWD;
    for (int p = t; p < HWD; p += 256) {
        int h = p / WWID, w = p % WWID;
        float s = 0.f;
#pragma unroll
        for (int i = 0; i < 7; ++i) {
            int hh = h + i - 3;
            if ((unsigned)hh < HH) {
#pragma unroll
                for (int j = 0; j < 7; ++j) {
                    int ww = w + j - 3;
                    if ((unsigned)ww < WWID) s += tile[hh * WWID + ww] * ker[i * 7 + j];
                }
            }
        }
        dst[p] = s;
    }
}

// -------------------------------------------------------------------------
// fp32 -> bf16 weight conversion
// -------------------------------------------------------------------------
__global__ void cvt_kernel(const float* __restrict__ src,
                           unsigned short* __restrict__ dst, int n) {
    int i = blockIdx.x * blockDim.x + threadIdx.x;
    if (i < n) dst[i] = f2bf(src[i]);
}

// -------------------------------------------------------------------------
extern "C" void kernel_launch(void* const* d_in, const int* in_sizes, int n_in,
                              void* d_out, int out_size, void* d_ws, size_t ws_size,
                              hipStream_t stream) {
    const float* x       = (const float*)d_in[0];
    /* d_in[1] = k (==7, compile-time) */
    const float* w_up    = (const float*)d_in[2];
    const float* b_up    = (const float*)d_in[3];
    const float* w_down  = (const float*)d_in[4];
    const float* b_down  = (const float*)d_in[5];
    const float* w_final = (const float*)d_in[6];
    const float* b_final = (const float*)d_in[7];
    float* outp = (float*)d_out;

    // workspace layout
    char* ws = (char*)d_ws;
    float* out_dw = (float*)ws;                                  // 205.5 MB
    size_t off = (size_t)NB * COUT * HWD * sizeof(float);
    float* pooled = (float*)(ws + off); off += (size_t)NB * CIN * PP * sizeof(float);
    float* down   = (float*)(ws + off); off += (size_t)NB * COUT * PP * sizeof(float);
    unsigned short* wup_bf  = (unsigned short*)(ws + off); off += (size_t)COUT * CIN * 2;
    unsigned short* wfin_bf = (unsigned short*)(ws + off);

    // 1) weights -> bf16
    cvt_kernel<<<(COUT * CIN + 255) / 256, 256, 0, stream>>>(w_up, wup_bf, COUT * CIN);
    cvt_kernel<<<(COUT * COUT + 255) / 256, 256, 0, stream>>>(w_final, wfin_bf, COUT * COUT);
    // 2) pool + down projection (dynamic depthwise kernels)
    pool_kernel<<<(NB * CIN * PP + 255) / 256, 256, 0, stream>>>(x, pooled);
    down_kernel<<<(NB * COUT * PP + 255) / 256, 256, 0, stream>>>(pooled, w_down, b_down, down);
    // 3) up = w_up @ x  (WMMA bf16; d_out used as scratch for `up`)
    dim3 ggrid(HWD / 64, COUT / 128, NB);
    gemm_wmma_bf16<<<ggrid, 128, 0, stream>>>(wup_bf, x, b_up, outp);
    // 4) dynamic depthwise conv
    dwconv_kernel<<<dim3(COUT, NB), 256, 0, stream>>>(outp, down, out_dw);
    // 5) final = w_final @ out
    gemm_wmma_bf16<<<ggrid, 128, 0, stream>>>(wfin_bf, out_dw, b_final, outp);
}
